// SimGCL_41987600286313
// MI455X (gfx1250) — compile-verified, matrix operand order checked by reading, not run on hardware
//
#include <hip/hip_runtime.h>
#include <math.h>

// ---- model hyperparams ----
#define K_USERS 200000
#define K_ITEMS 100000
#define K_N     300000          // K_USERS + K_ITEMS
#define K_DIM   64
#define K_NNZ   3000000
#define K_BATCH 4096
#define K_HOPS  3
#define K_EPS   0.1f
#define K_TEMP  0.2f
#define K_CLR   0.5f
#define K_DECAY 1e-4f

typedef __attribute__((ext_vector_type(16))) _Float16 v16h;
typedef __attribute__((ext_vector_type(8)))  _Float16 v8h;
typedef __attribute__((ext_vector_type(8)))  float    v8f;

__device__ __forceinline__ float atomAddF(float* p, float v) {
  return __hip_atomic_fetch_add(p, v, __ATOMIC_RELAXED, __HIP_MEMORY_SCOPE_AGENT);
}

__device__ __forceinline__ unsigned pcg(unsigned x) {
  x = x * 747796405u + 2891336453u;
  unsigned w = ((x >> ((x >> 28u) + 4u)) ^ x) * 277803737u;
  return (w >> 22u) ^ w;
}

__device__ __forceinline__ float waveSum32(float v) {
#pragma unroll
  for (int off = 16; off > 0; off >>= 1) v += __shfl_xor(v, off);
  return v;
}

// ---------------- utility kernels ----------------
__global__ void k_zero4(float4* __restrict__ p, int n4) {
  int i = blockIdx.x * blockDim.x + threadIdx.x;
  if (i < n4) p[i] = float4{0.f, 0.f, 0.f, 0.f};
}

__global__ void k_zerof(float* __restrict__ p, int n) {
  int i = blockIdx.x * blockDim.x + threadIdx.x;
  if (i < n) p[i] = 0.f;
}

// E = concat(user_embed, item_embed), float4 vectorized
__global__ void k_concat4(const float4* __restrict__ ue, const float4* __restrict__ ie,
                          float4* __restrict__ E, int n4u, int n4t) {
  int i = blockIdx.x * blockDim.x + threadIdx.x;
  if (i < n4t) E[i] = (i < n4u) ? ue[i] : ie[i - n4u];
}

// ---------------- SPMM: Y[r] += v * X[c], COO ----------------
// 16 threads per nonzero, each handles 4 contiguous dims (float4 gather,
// 4x f32 global atomics).  X and Y (77MB each) are L2-resident on MI455X.
__global__ void __launch_bounds__(256)
k_spmm(const float* __restrict__ vals, const int* __restrict__ rows,
       const int* __restrict__ cols, const float* __restrict__ X,
       float* __restrict__ Y) {
  int t = blockIdx.x * blockDim.x + threadIdx.x;
  if (t >= K_NNZ * 16) return;
  int e = t >> 4;
  int q = t & 15;
  int r = rows[e];
  int c = cols[e];
  float v = vals[e];
  float4 x = *(const float4*)(X + (size_t)c * K_DIM + q * 4);
  float* yp = Y + (size_t)r * K_DIM + q * 4;
  atomAddF(yp + 0, v * x.x);
  atomAddF(yp + 1, v * x.y);
  atomAddF(yp + 2, v * x.z);
  atomAddF(yp + 3, v * x.w);
}

// ---------------- per-hop perturb (SimGCL noise) + accumulate ----------------
// One wave32 per row; lane handles 2 dims.  Row-normalized hash noise.
__global__ void __launch_bounds__(256)
k_perturb_acc(float* __restrict__ E, float* __restrict__ ACC,
              int perturbed, unsigned seed, int hop) {
  int wave = (blockIdx.x * blockDim.x + threadIdx.x) >> 5;
  int lane = threadIdx.x & 31;
  if (wave >= K_N) return;
  int base = wave * K_DIM + lane * 2;
  float e0 = E[base], e1 = E[base + 1];
  if (perturbed) {
    unsigned key = pcg(seed * 0x9E3779B1u + (unsigned)hop * 0x7F4A7C15u);
    float n0 = (pcg(key ^ (unsigned)base) >> 8) * (1.0f / 16777216.0f);
    float n1 = (pcg(key ^ (unsigned)(base + 1)) >> 8) * (1.0f / 16777216.0f);
    float nrm = waveSum32(n0 * n0 + n1 * n1);
    nrm = fmaxf(sqrtf(nrm), 1e-12f);
    float s0 = (e0 > 0.f) ? 1.f : ((e0 < 0.f) ? -1.f : 0.f);
    float s1 = (e1 > 0.f) ? 1.f : ((e1 < 0.f) ? -1.f : 0.f);
    e0 += s0 * (n0 / nrm) * K_EPS;
    e1 += s1 * (n1 / nrm) * K_EPS;
    E[base] = e0;
    E[base + 1] = e1;
  }
  ACC[base] += e0;
  ACC[base + 1] += e1;
}

// ---------------- gathers ----------------
// rec branch: ue/pe/ne = mean rows (un-normalized), /= n_hops
__global__ void k_gather_rec(const float* __restrict__ ACC,
                             const int* __restrict__ users,
                             const int* __restrict__ pos,
                             const int* __restrict__ neg,
                             float* __restrict__ ue, float* __restrict__ pe,
                             float* __restrict__ ne) {
  int t = blockIdx.x * blockDim.x + threadIdx.x;
  if (t >= K_BATCH * 16) return;
  int b = t >> 4, q = (t & 15) * 4;
  const float inv = 1.0f / K_HOPS;
  float4 a = *(const float4*)(ACC + (size_t)users[b] * K_DIM + q);
  *(float4*)(ue + b * K_DIM + q) = float4{a.x * inv, a.y * inv, a.z * inv, a.w * inv};
  a = *(const float4*)(ACC + (size_t)(K_USERS + pos[b]) * K_DIM + q);
  *(float4*)(pe + b * K_DIM + q) = float4{a.x * inv, a.y * inv, a.z * inv, a.w * inv};
  a = *(const float4*)(ACC + (size_t)(K_USERS + neg[b]) * K_DIM + q);
  *(float4*)(ne + b * K_DIM + q) = float4{a.x * inv, a.y * inv, a.z * inv, a.w * inv};
}

// CL branch: gather rows, L2-normalize, store f16 (WMMA operands).
__global__ void __launch_bounds__(256)
k_gather_norm_h(const float* __restrict__ ACC, const int* __restrict__ idx,
                int rowbase, _Float16* __restrict__ out) {
  int wave = (blockIdx.x * blockDim.x + threadIdx.x) >> 5;
  int lane = threadIdx.x & 31;
  if (wave >= K_BATCH) return;
  size_t sbase = (size_t)(rowbase + idx[wave]) * K_DIM + lane * 2;
  float a0 = ACC[sbase] * (1.0f / K_HOPS);
  float a1 = ACC[sbase + 1] * (1.0f / K_HOPS);
  float nrm = waveSum32(a0 * a0 + a1 * a1);
  nrm = fmaxf(sqrtf(nrm), 1e-12f);
  out[wave * K_DIM + lane * 2]     = (_Float16)(a0 / nrm);
  out[wave * K_DIM + lane * 2 + 1] = (_Float16)(a1 / nrm);
}

// ---------------- rec loss + reg (Frobenius^2 partials) ----------------
__global__ void __launch_bounds__(256)
k_rec_loss(const float* __restrict__ ue, const float* __restrict__ pe,
           const float* __restrict__ ne, float* __restrict__ accum) {
  int wave = (blockIdx.x * blockDim.x + threadIdx.x) >> 5;
  int lane = threadIdx.x & 31;
  if (wave >= K_BATCH) return;
  int base = wave * K_DIM + lane * 2;
  float u0 = ue[base], u1 = ue[base + 1];
  float p0 = pe[base], p1 = pe[base + 1];
  float n0 = ne[base], n1 = ne[base + 1];
  float pos = waveSum32(u0 * p0 + u1 * p1);
  float neg = waveSum32(u0 * n0 + u1 * n1);
  float usq = waveSum32(u0 * u0 + u1 * u1);
  float psq = waveSum32(p0 * p0 + p1 * p1);
  if (lane == 0) {
    float sig = 1.0f / (1.0f + __expf(-(pos - neg)));
    atomAddF(&accum[0], -__logf(1e-5f + sig));
    atomAddF(&accum[1], usq);
    atomAddF(&accum[2], psq);
  }
}

// ---------------- InfoNCE GEMM via WMMA ----------------
// One wave per 16x16 tile of exp((V1 @ V2^T)/temp); K=64 done as two
// v_wmma_f32_16x16x32_f16 steps; row sums -> ttl via cross-lane reduce +
// one atomic per row-half; diagonal tiles also emit pos[].
__global__ void __launch_bounds__(256)
k_nce_gemm(const _Float16* __restrict__ V1, const _Float16* __restrict__ V2,
           float* __restrict__ ttl, float* __restrict__ pos) {
  int ti = blockIdx.x;                    // row tile (0..255)
  int tj = blockIdx.y * 8 + (threadIdx.x >> 5);   // col tile (0..255)
  int lane = threadIdx.x & 31;
  int half = lane >> 4;                   // K sub-block select
  int l16 = lane & 15;

  const _Float16* arow = V1 + (size_t)(ti * 16 + l16) * K_DIM;
  const _Float16* brow = V2 + (size_t)(tj * 16 + l16) * K_DIM;

  v8f c = {};
#pragma unroll
  for (int kc = 0; kc < 2; ++kc) {
    int kb = kc * 32 + half * 8;          // lane's K base within the chunk
    v8h alo = *(const v8h*)(arow + kb);
    v8h ahi = *(const v8h*)(arow + kb + 16);
    v8h blo = *(const v8h*)(brow + kb);
    v8h bhi = *(const v8h*)(brow + kb + 16);
    v16h a = __builtin_shufflevector(alo, ahi, 0, 1, 2, 3, 4, 5, 6, 7,
                                     8, 9, 10, 11, 12, 13, 14, 15);
    v16h b = __builtin_shufflevector(blo, bhi, 0, 1, 2, 3, 4, 5, 6, 7,
                                     8, 9, 10, 11, 12, 13, 14, 15);
    c = __builtin_amdgcn_wmma_f32_16x16x32_f16(false, a, false, b,
                                               (short)0, c, false, false);
  }

  // C layout: VGPR r -> row M = r + 8*half, col N = l16
#pragma unroll
  for (int r = 0; r < 8; ++r) {
    int M = r + 8 * half;
    float ev = __expf(c[r] * (1.0f / K_TEMP));
    float s = ev;
#pragma unroll
    for (int off = 1; off < 16; off <<= 1) s += __shfl_xor(s, off);
    if (l16 == 0) atomAddF(&ttl[ti * 16 + M], s);
    if (ti == tj && l16 == M) pos[ti * 16 + M] = ev;
  }
}

__global__ void __launch_bounds__(256)
k_nce_final(const float* __restrict__ pos, const float* __restrict__ ttl,
            float* __restrict__ accum, int slot) {
  int b = blockIdx.x * blockDim.x + threadIdx.x;
  float v = 0.f;
  if (b < K_BATCH) v = -__logf(pos[b] / ttl[b] + 1e-5f);
  v = waveSum32(v);
  if ((threadIdx.x & 31) == 0) atomAddF(&accum[slot], v);
}

__global__ void k_final(const float* __restrict__ accum, float* __restrict__ out) {
  if (blockIdx.x == 0 && threadIdx.x == 0) {
    float rec = accum[0] * (1.0f / K_BATCH);
    float reg = K_DECAY * (sqrtf(accum[1]) + sqrtf(accum[2]));
    float cl  = K_CLR * (accum[3] * (1.0f / K_BATCH) + accum[4] * (1.0f / K_BATCH));
    out[0] = rec + reg + cl;
  }
}

// ---------------- host orchestration ----------------
static void gcn_run(const float* u_emb, const float* i_emb,
                    const float* vals, const int* rows, const int* cols,
                    float* E, float* T, float* ACC,
                    int perturbed, unsigned seed, hipStream_t stream) {
  const int n4t = K_N * K_DIM / 4;
  const int n4u = K_USERS * K_DIM / 4;
  const int zb = (n4t + 255) / 256;
  k_zero4<<<zb, 256, 0, stream>>>((float4*)ACC, n4t);
  k_concat4<<<zb, 256, 0, stream>>>((const float4*)u_emb, (const float4*)i_emb,
                                    (float4*)E, n4u, n4t);
  float* X = E;
  float* Y = T;
  const int spmm_blocks = (K_NNZ * 16 + 255) / 256;
  const int pa_blocks = (K_N * 32 + 255) / 256;
  for (int hop = 0; hop < K_HOPS; ++hop) {
    k_zero4<<<zb, 256, 0, stream>>>((float4*)Y, n4t);
    k_spmm<<<spmm_blocks, 256, 0, stream>>>(vals, rows, cols, X, Y);
    k_perturb_acc<<<pa_blocks, 256, 0, stream>>>(Y, ACC, perturbed, seed, hop);
    float* tmp = X; X = Y; Y = tmp;
  }
}

extern "C" void kernel_launch(void* const* d_in, const int* in_sizes, int n_in,
                              void* d_out, int out_size, void* d_ws, size_t ws_size,
                              hipStream_t stream) {
  const float* user_embed = (const float*)d_in[0];
  const float* item_embed = (const float*)d_in[1];
  const float* adj_vals   = (const float*)d_in[2];
  const int*   adj_rows   = (const int*)d_in[3];
  const int*   adj_cols   = (const int*)d_in[4];
  const int*   users      = (const int*)d_in[5];
  const int*   pos_items  = (const int*)d_in[6];
  const int*   neg_items  = (const int*)d_in[7];

  char* ws = (char*)d_ws;
  size_t off = 0;
  auto alloc = [&](size_t bytes) -> char* {
    char* p = ws + off;
    off += (bytes + 255) & ~(size_t)255;
    return p;
  };
  float* E   = (float*)alloc(sizeof(float) * (size_t)K_N * K_DIM);
  float* T   = (float*)alloc(sizeof(float) * (size_t)K_N * K_DIM);
  float* ACC = (float*)alloc(sizeof(float) * (size_t)K_N * K_DIM);
  float* ue = (float*)alloc(sizeof(float) * K_BATCH * K_DIM);
  float* pe = (float*)alloc(sizeof(float) * K_BATCH * K_DIM);
  float* ne = (float*)alloc(sizeof(float) * K_BATCH * K_DIM);
  _Float16* u1h = (_Float16*)alloc(sizeof(_Float16) * K_BATCH * K_DIM);
  _Float16* u2h = (_Float16*)alloc(sizeof(_Float16) * K_BATCH * K_DIM);
  _Float16* i1h = (_Float16*)alloc(sizeof(_Float16) * K_BATCH * K_DIM);
  _Float16* i2h = (_Float16*)alloc(sizeof(_Float16) * K_BATCH * K_DIM);
  float* ttl_u = (float*)alloc(sizeof(float) * K_BATCH);
  float* ttl_i = (float*)alloc(sizeof(float) * K_BATCH);
  float* pos_u = (float*)alloc(sizeof(float) * K_BATCH);
  float* pos_i = (float*)alloc(sizeof(float) * K_BATCH);
  float* accum = (float*)alloc(sizeof(float) * 8);

  // zero scalar accumulators + ttl sums (we accumulate atomically)
  k_zerof<<<1, 64, 0, stream>>>(accum, 8);
  k_zerof<<<(K_BATCH + 255) / 256, 256, 0, stream>>>(ttl_u, K_BATCH);
  k_zerof<<<(K_BATCH + 255) / 256, 256, 0, stream>>>(ttl_i, K_BATCH);

  const int wave_blocks = (K_BATCH * 32 + 255) / 256;

  // --- recommendation branch (no perturbation) ---
  gcn_run(user_embed, item_embed, adj_vals, adj_rows, adj_cols,
          E, T, ACC, /*perturbed=*/0, 0u, stream);
  k_gather_rec<<<(K_BATCH * 16 + 255) / 256, 256, 0, stream>>>(
      ACC, users, pos_items, neg_items, ue, pe, ne);
  k_rec_loss<<<wave_blocks, 256, 0, stream>>>(ue, pe, ne, accum);

  // --- contrastive view 1 ---
  gcn_run(user_embed, item_embed, adj_vals, adj_rows, adj_cols,
          E, T, ACC, /*perturbed=*/1, 101u, stream);
  k_gather_norm_h<<<wave_blocks, 256, 0, stream>>>(ACC, users, 0, u1h);
  k_gather_norm_h<<<wave_blocks, 256, 0, stream>>>(ACC, pos_items, K_USERS, i1h);

  // --- contrastive view 2 ---
  gcn_run(user_embed, item_embed, adj_vals, adj_rows, adj_cols,
          E, T, ACC, /*perturbed=*/1, 202u, stream);
  k_gather_norm_h<<<wave_blocks, 256, 0, stream>>>(ACC, users, 0, u2h);
  k_gather_norm_h<<<wave_blocks, 256, 0, stream>>>(ACC, pos_items, K_USERS, i2h);

  // --- InfoNCE: WMMA tiled exp-similarity row sums ---
  dim3 ggrid(K_BATCH / 16, (K_BATCH / 16) / 8);  // 256 x 32 blocks, 8 waves/block
  k_nce_gemm<<<ggrid, 256, 0, stream>>>(u1h, u2h, ttl_u, pos_u);
  k_nce_gemm<<<ggrid, 256, 0, stream>>>(i1h, i2h, ttl_i, pos_i);
  k_nce_final<<<(K_BATCH + 255) / 256, 256, 0, stream>>>(pos_u, ttl_u, accum, 3);
  k_nce_final<<<(K_BATCH + 255) / 256, 256, 0, stream>>>(pos_i, ttl_i, accum, 4);

  k_final<<<1, 64, 0, stream>>>(accum, (float*)d_out);
}
